// CodeBook_32607391711268
// MI455X (gfx1250) — compile-verified, matrix operand order checked by reading, not run on hardware
//
#include <hip/hip_runtime.h>

// ---------------------------------------------------------------------------
// VQ nearest-codebook search on gfx1250 (MI455X)
//   scores S = X (M x 128, f16) @ C^T (128 x 1024, f16) via v_wmma_f32_16x16x32_f16
//   argmin ||x-c||^2  ==  argmax (x.c - 0.5*||c||^2)
//   - codebook chunk (128 KB f16) + 0.5*||c||^2 (4 KB) staged in LDS
//   - each wave owns TWO 16-row A tiles -> every B fragment feeds 2 WMMAs
//   outputs: gathered exact f32 codebook rows + indices (as float)
// ---------------------------------------------------------------------------

typedef _Float16 v16h __attribute__((ext_vector_type(16)));
typedef _Float16 h8   __attribute__((ext_vector_type(8)));
typedef _Float16 h4   __attribute__((ext_vector_type(4)));
typedef float    v8f  __attribute__((ext_vector_type(8)));
typedef float    f4   __attribute__((ext_vector_type(4)));

#define DIM           128     // vector dim
#define KENT          1024    // codebook entries
#define CHUNK         512     // codebook entries staged in LDS per pass (128 KB f16)
#define WAVES         8
#define BLOCK         (WAVES * 32)
#define ROWS_PER_WAVE 32      // two 16-row WMMA tiles per wave
#define ROWS_PER_WG   (WAVES * ROWS_PER_WAVE)   // 256

extern __shared__ _Float16 lds_cb[];   // [CHUNK*DIM halves][KENT floats of 0.5*csq]

// ---- prep: f32 codebook -> f16 copy in ws, plus ||c||^2 per entry ---------
__global__ __launch_bounds__(32) void vq_prep(const float* __restrict__ cbf,
                                              _Float16* __restrict__ cb16,
                                              float* __restrict__ csq) {
    const int k    = blockIdx.x;      // one wave per codebook row
    const int lane = threadIdx.x;     // 0..31, each lane owns 4 dims
    f4 v = *(const f4*)(cbf + (size_t)k * DIM + lane * 4);
    h4 h;
#pragma unroll
    for (int i = 0; i < 4; ++i) h[i] = (_Float16)v[i];
    *(h4*)(cb16 + (size_t)k * DIM + lane * 4) = h;
    float s = v[0]*v[0] + v[1]*v[1] + v[2]*v[2] + v[3]*v[3];
#pragma unroll
    for (int m = 1; m < 32; m <<= 1) s += __shfl_xor(s, m, 32);
    if (lane == 0) csq[k] = s;
}

// ---- main: per wave, 32 query rows vs all 1024 entries --------------------
__global__ __launch_bounds__(BLOCK) void vq_main(const float* __restrict__ codes,
                                                 const float* __restrict__ cbf,
                                                 const _Float16* __restrict__ cb16,
                                                 const float* __restrict__ csq,
                                                 float* __restrict__ out_codes,
                                                 float* __restrict__ out_idx) {
    const int tid       = threadIdx.x;
    const int lane      = tid & 31;
    const int wave      = tid >> 5;
    const int rows_base = blockIdx.x * ROWS_PER_WG + wave * ROWS_PER_WAVE;
    const int m         = lane & 15;            // row (A) / column (B,C) slot
    const int sel       = (lane < 16) ? 0 : 8;  // K-half selector per ISA layout

    float* lds_csq = (float*)(lds_cb + (size_t)CHUNK * DIM);

    // A fragments for two 16-row tiles (documented 16-bit A 16x32 layout:
    // lane<16: K 0..7,16..23 ; lane>=16: K 8..15,24..31), streamed non-temporal.
    v16h afrag0[4], afrag1[4];
#pragma unroll
    for (int t = 0; t < 2; ++t) {
        const float* arow = codes + (size_t)(rows_base + t * 16 + m) * DIM;
#pragma unroll
        for (int kt = 0; kt < 4; ++kt) {
            const float* s0 = arow + kt * 32 + sel;
            f4 x0 = __builtin_nontemporal_load((const f4*)(s0));
            f4 x1 = __builtin_nontemporal_load((const f4*)(s0 + 4));
            f4 x2 = __builtin_nontemporal_load((const f4*)(s0 + 16));
            f4 x3 = __builtin_nontemporal_load((const f4*)(s0 + 20));
            v16h a;
#pragma unroll
            for (int i = 0; i < 4; ++i) {
                a[i]      = (_Float16)x0[i];
                a[4 + i]  = (_Float16)x1[i];
                a[8 + i]  = (_Float16)x2[i];
                a[12 + i] = (_Float16)x3[i];
            }
            if (t == 0) afrag0[kt] = a; else afrag1[kt] = a;
        }
    }

    // stage 0.5*||c||^2 for all 1024 entries into LDS (covered by 1st barrier)
    for (int i = tid; i < KENT; i += BLOCK) lds_csq[i] = 0.5f * csq[i];

    float best0[8], best1[8];
    int   bestn0[8], bestn1[8];
#pragma unroll
    for (int r = 0; r < 8; ++r) {
        best0[r] = -3.0e38f; bestn0[r] = 0;
        best1[r] = -3.0e38f; bestn1[r] = 0;
    }

    for (int chunk = 0; chunk < KENT / CHUNK; ++chunk) {
        // cooperative stage: 128 KB f16 chunk -> LDS (each thread 32 x b128)
        {
            const uint4* src = (const uint4*)(cb16 + (size_t)chunk * CHUNK * DIM);
            uint4*       dst = (uint4*)lds_cb;
#pragma unroll 4
            for (int i = tid; i < CHUNK * DIM / 8; i += BLOCK) dst[i] = src[i];
        }
        __syncthreads();

#pragma unroll 2
        for (int nt = 0; nt < CHUNK / 16; ++nt) {
            const int n = chunk * CHUNK + nt * 16 + m;   // this lane's column
            v8f acc0 = {}, acc1 = {};
#pragma unroll
            for (int kt = 0; kt < 4; ++kt) {
                // B fragment (32x16, mirrored layout): entry = column, dims = K
                const _Float16* bsrc = lds_cb + (size_t)(nt * 16 + m) * DIM + kt * 32 + sel;
                h8 b0 = *(const h8*)(bsrc);
                h8 b1 = *(const h8*)(bsrc + 16);
                v16h b = __builtin_shufflevector(b0, b1,
                        0,1,2,3,4,5,6,7,8,9,10,11,12,13,14,15);
                acc0 = __builtin_amdgcn_wmma_f32_16x16x32_f16(
                        false, afrag0[kt], false, b, (short)0, acc0, false, false);
                acc1 = __builtin_amdgcn_wmma_f32_16x16x32_f16(
                        false, afrag1[kt], false, b, (short)0, acc1, false, false);
            }
            const float hc = lds_csq[n];                 // 0.5*||c||^2 from LDS
#pragma unroll
            for (int r = 0; r < 8; ++r) {
                float v0 = acc0[r] - hc;                 // argmax(dot - 0.5*||c||^2)
                float v1 = acc1[r] - hc;
                if (v0 > best0[r]) { best0[r] = v0; bestn0[r] = n; }
                if (v1 > best1[r]) { best1[r] = v1; bestn1[r] = n; }
            }
        }
        __syncthreads();
    }

    // reduce across the 16 lanes that share each output row (xor tree stays
    // inside each 16-lane half of the wave32); tie-break to lowest index
#pragma unroll
    for (int msk = 1; msk <= 8; msk <<= 1) {
#pragma unroll
        for (int r = 0; r < 8; ++r) {
            float ov0 = __shfl_xor(best0[r], msk, 32);
            int   oi0 = __shfl_xor(bestn0[r], msk, 32);
            if (ov0 > best0[r] || (ov0 == best0[r] && oi0 < bestn0[r])) {
                best0[r] = ov0; bestn0[r] = oi0;
            }
            float ov1 = __shfl_xor(best1[r], msk, 32);
            int   oi1 = __shfl_xor(bestn1[r], msk, 32);
            if (ov1 > best1[r] || (ov1 == best1[r] && oi1 < bestn1[r])) {
                best1[r] = ov1; bestn1[r] = oi1;
            }
        }
    }

    // lane 0 holds rows +0..7 / +16..23 ; lane 16 holds rows +8..15 / +24..31
    if ((lane & 15) == 0) {
        const int roff = (lane >> 4) * 8;
#pragma unroll
        for (int r = 0; r < 8; ++r) {
            out_idx[rows_base + roff + r]      = (float)bestn0[r];
            out_idx[rows_base + 16 + roff + r] = (float)bestn1[r];
        }
    }

    // gather winning rows from the exact f32 codebook: 32 lanes x float4 = 512B/row
#pragma unroll
    for (int r = 0; r < 16; ++r) {
        const int srcLane = (r < 8) ? 0 : 16;
        const int k0 = __shfl(bestn0[r & 7], srcLane, 32);
        const int k1 = __shfl(bestn1[r & 7], srcLane, 32);
        f4 v0 = *(const f4*)(cbf + (size_t)k0 * DIM + lane * 4);
        f4 v1 = *(const f4*)(cbf + (size_t)k1 * DIM + lane * 4);
        __builtin_nontemporal_store(v0,
            (f4*)(out_codes + (size_t)(rows_base + r) * DIM + lane * 4));
        __builtin_nontemporal_store(v1,
            (f4*)(out_codes + (size_t)(rows_base + 16 + r) * DIM + lane * 4));
    }
}

extern "C" void kernel_launch(void* const* d_in, const int* in_sizes, int n_in,
                              void* d_out, int out_size, void* d_ws, size_t ws_size,
                              hipStream_t stream) {
    const float* codes = (const float*)d_in[0];   // (B, N, 128) f32
    const float* cbf   = (const float*)d_in[1];   // (1024, 128) f32, rows normalized

    // ws layout: [f16 codebook 256 KB][csq 4 KB]
    _Float16* cb16 = (_Float16*)d_ws;
    float*    csq  = (float*)((char*)d_ws + (size_t)KENT * DIM * sizeof(_Float16));

    const int M = in_sizes[0] / DIM;              // 131072 query vectors
    float* out_codes = (float*)d_out;             // M*128 floats
    float* out_idx   = out_codes + (size_t)M * DIM; // M floats (indices)

    const size_t lds_bytes = (size_t)CHUNK * DIM * sizeof(_Float16)
                           + (size_t)KENT * sizeof(float);

    vq_prep<<<KENT, 32, 0, stream>>>(cbf, cb16, csq);
    vq_main<<<M / ROWS_PER_WG, BLOCK, lds_bytes, stream>>>(
        codes, cbf, cb16, csq, out_codes, out_idx);
}
